// RelationalGNNLayer_23845658427999
// MI455X (gfx1250) — compile-verified, matrix operand order checked by reading, not run on hardware
//
#include <hip/hip_runtime.h>
#include <hip/hip_bf16.h>
#include <stdint.h>

// ---------------------------------------------------------------------------
// RelationalGNNLayer for MI455X (gfx1250): bf16 WMMA 16x16x32, f32 accum.
// B=8, K=2048, D=256 hardcoded from the reference.
// Async LDS staging (double-buffered) of the B slab in the big GEMM,
// packed bf16 conversion via v_cvt_pk_bf16_f32 when available.
// ---------------------------------------------------------------------------

typedef __attribute__((ext_vector_type(16))) __bf16 v16bf;
typedef __attribute__((ext_vector_type(2)))  __bf16 v2bf;
typedef __attribute__((ext_vector_type(8)))  float  v8f;
typedef __attribute__((ext_vector_type(4)))  int    v4i;

// pointer-to-int4 in global (AS1) / LDS (AS3) address spaces, as required by
// __builtin_amdgcn_global_load_async_to_lds_b128 (param 1: "int4 __device__*")
typedef __attribute__((address_space(1))) v4i* as1_v4i_p;
typedef __attribute__((address_space(3))) v4i* as3_v4i_p;

#define L2E 1.44269504088896340736f

#if __has_builtin(__builtin_amdgcn_exp2f)
#define EXP2F(x) __builtin_amdgcn_exp2f(x)
#else
#define EXP2F(x) exp2f(x)
#endif

#if __has_builtin(__builtin_amdgcn_global_load_async_to_lds_b128) && \
    __has_builtin(__builtin_amdgcn_s_wait_asynccnt)
#define USE_ASYNC_LDS 1
#else
#define USE_ASYNC_LDS 0
#endif

static __device__ __forceinline__ unsigned short f2bf(float f) {
  unsigned int u = __builtin_bit_cast(unsigned int, f);
  unsigned int r = u + 0x7FFFu + ((u >> 16) & 1u);   // round-to-nearest-even
  return (unsigned short)(r >> 16);
}

// pack two floats -> two bf16 in one dword (v_cvt_pk_bf16_f32 when present)
static __device__ __forceinline__ unsigned int f2bf2(float lo, float hi) {
#if __has_builtin(__builtin_amdgcn_cvt_pk_bf16_f32)
  v2bf r = __builtin_amdgcn_cvt_pk_bf16_f32(lo, hi);
  return __builtin_bit_cast(unsigned int, r);
#else
  return (unsigned int)f2bf(lo) | ((unsigned int)f2bf(hi) << 16);
#endif
}

union ABf {                    // one 16x32 (or 32x16) bf16 WMMA fragment / lane
  v16bf v;
  unsigned short s[16];
  unsigned int u[8];
  uint4 q[2];
};
union Pack8 { unsigned int u32[4]; uint4 u; };
union Pack4 { unsigned int u32[2]; uint2 u; };

static __device__ __forceinline__ void cvt4(const float4& x, unsigned int* d) {
  d[0] = f2bf2(x.x, x.y);
  d[1] = f2bf2(x.z, x.w);
}
static __device__ __forceinline__ void cvt4p(const float4& x, float rm, float is,
                                             unsigned int* d) {
  const float p0 = EXP2F((x.x - rm) * L2E) * is;
  const float p1 = EXP2F((x.y - rm) * L2E) * is;
  const float p2 = EXP2F((x.z - rm) * L2E) * is;
  const float p3 = EXP2F((x.w - rm) * L2E) * is;
  d[0] = f2bf2(p0, p1);
  d[1] = f2bf2(p2, p3);
}

// ---------------------------------------------------------------------------
// Kernel 0: bf16 transposed weight copies (B-fragment friendly layouts).
// ---------------------------------------------------------------------------
__global__ void __launch_bounds__(256) prep_weights(
    const float* __restrict__ msg_W, const float* __restrict__ W1,
    const float* __restrict__ W2, unsigned short* __restrict__ msgWT,
    unsigned short* __restrict__ W1T, unsigned short* __restrict__ W2T) {
  int id = blockIdx.x * 256 + threadIdx.x;
  if (id < 65536) {
    int n = id >> 8, k = id & 255;
    msgWT[id] = f2bf(msg_W[k * 256 + n]);
  } else if (id < 65536 + 131072) {
    int j = id - 65536;
    int n = j >> 9, c = j & 511;
    W1T[j] = f2bf(W1[c * 256 + n]);
  } else if (id < 262144) {
    int j = id - 196608;
    int n = j >> 8, d = j & 255;
    W2T[j] = f2bf(W2[d * 256 + n]);
  }
}

// ---------------------------------------------------------------------------
// Kernel 1: messages = V @ msg_W + msg_b, stored TRANSPOSED bf16 [B][256][2048]
// ---------------------------------------------------------------------------
__global__ void __launch_bounds__(256) msg_kernel(
    const float* __restrict__ V, const unsigned short* __restrict__ msgWT,
    const float* __restrict__ msg_b, unsigned short* __restrict__ msgsT) {
  const int R0 = blockIdx.x * 64;         // global row over B*K (64 | 2048)
  const int b = R0 >> 11;
  const int rloc0 = R0 & 2047;
  const int w = threadIdx.x >> 5, lane = threadIdx.x & 31;
  const int mi = w >> 1, nh = w & 1;
  const int m = lane & 15, hi = lane >> 4;
  const float* rp = V + (size_t)(R0 + mi * 16 + m) * 256;

  v8f acc[8] = {};
  for (int kk = 0; kk < 256; kk += 32) {
    const int c0 = kk + hi * 8, c1 = c0 + 16;
    ABf a;
    cvt4(*(const float4*)(rp + c0),     a.u + 0);
    cvt4(*(const float4*)(rp + c0 + 4), a.u + 2);
    cvt4(*(const float4*)(rp + c1),     a.u + 4);
    cvt4(*(const float4*)(rp + c1 + 4), a.u + 6);
#pragma unroll
    for (int t = 0; t < 8; ++t) {
      const int n = nh * 128 + t * 16 + m;
      const unsigned short* bp = msgWT + (size_t)n * 256;
      ABf bf;
      bf.q[0] = *(const uint4*)(bp + c0);
      bf.q[1] = *(const uint4*)(bp + c1);
      acc[t] = __builtin_amdgcn_wmma_f32_16x16x32_bf16(
          false, a.v, false, bf.v, (short)0, acc[t], false, false);
    }
  }
#pragma unroll
  for (int t = 0; t < 8; ++t) {
    const int n = nh * 128 + t * 16 + m;
    const float bias = msg_b[n];
    Pack8 p;
#pragma unroll
    for (int i = 0; i < 4; ++i)
      p.u32[i] = f2bf2(acc[t][2 * i] + bias, acc[t][2 * i + 1] + bias);
    const int rloc = rloc0 + mi * 16 + hi * 8;           // 8 contiguous k's
    *(uint4*)(msgsT + (size_t)(b * 256 + n) * 2048 + rloc) = p.u;
  }
}

// ---------------------------------------------------------------------------
// Kernel 2: per-row softmax stats (online max / sum of exp), 1 wave per row.
// ---------------------------------------------------------------------------
__global__ void __launch_bounds__(256) softmax_stats(
    const float* __restrict__ E, float* __restrict__ rowmax,
    float* __restrict__ invsum) {
  const int w = threadIdx.x >> 5, lane = threadIdx.x & 31;
  const int row = blockIdx.x * 8 + w;            // over B*K = 16384
  const float* er = E + (size_t)row * 2048;
  float m = -3.0e38f, s = 0.0f;
  for (int i = 0; i < 16; ++i) {
    const float4 x = *(const float4*)(er + lane * 4 + i * 128);
    const float xs[4] = {x.x, x.y, x.z, x.w};
#pragma unroll
    for (int j = 0; j < 4; ++j) {
      const float nm = fmaxf(m, xs[j]);
      s = s * EXP2F((m - nm) * L2E) + EXP2F((xs[j] - nm) * L2E);
      m = nm;
    }
  }
#pragma unroll
  for (int off = 16; off >= 1; off >>= 1) {
    const float om = __shfl_xor(m, off, 32);
    const float os = __shfl_xor(s, off, 32);
    const float nm = fmaxf(m, om);
    s = s * EXP2F((m - nm) * L2E) + os * EXP2F((om - nm) * L2E);
    m = nm;
  }
  if (lane == 0) { rowmax[row] = m; invsum[row] = 1.0f / s; }
}

// ---------------------------------------------------------------------------
// Kernel 3: agg = softmax(E) @ messages, fused (P built on the fly in bf16).
// Block: 8 waves -> 64 rows x 256 cols of output, K-loop 2048 in steps of 32.
// B slab (32x256 bf16 = 16KB/step) staged into LDS via async DMA, double-
// buffered: DMA of step it+1 overlaps WMMA of step it.
// ---------------------------------------------------------------------------
__global__ void __launch_bounds__(256) attn_kernel(
    const float* __restrict__ E, const unsigned short* __restrict__ msgsT,
    const float* __restrict__ rowmax, const float* __restrict__ invsum,
    float* __restrict__ agg) {
#if USE_ASYNC_LDS
  __shared__ unsigned short ldsB[2][256 * 32];   // [stage][n*32 + k]
#endif
  const int b = blockIdx.x >> 5;
  const int r0 = (blockIdx.x & 31) * 64;
  const int tid = threadIdx.x;
  const int w = tid >> 5, lane = tid & 31;
  const int mi = w >> 1, nh = w & 1;
  const int m = lane & 15, hi = lane >> 4;
  const int grow = r0 + mi * 16 + m;
  const float rm = rowmax[b * 2048 + grow];
  const float is = invsum[b * 2048 + grow];
  const float* er = E + ((size_t)b * 2048 + grow) * 2048;
  const unsigned short* mb = msgsT + (size_t)b * 256 * 2048;

#if USE_ASYNC_LDS
  // thread tid cooperatively copies messages row n = tid (64B per k-step)
  const unsigned short* gsrc = mb + (size_t)tid * 2048;
#define ISSUE_B(kk_, stage_)                                                  \
  do {                                                                        \
    const unsigned short* gp_ = gsrc + (kk_);                                 \
    unsigned short* lp_ = &ldsB[(stage_)][tid * 32];                          \
    __builtin_amdgcn_global_load_async_to_lds_b128(                           \
        (as1_v4i_p)(gp_),      (as3_v4i_p)(lp_), 0, 0);                       \
    __builtin_amdgcn_global_load_async_to_lds_b128(                           \
        (as1_v4i_p)(gp_ + 8),  (as3_v4i_p)(lp_ + 8), 0, 0);                   \
    __builtin_amdgcn_global_load_async_to_lds_b128(                           \
        (as1_v4i_p)(gp_ + 16), (as3_v4i_p)(lp_ + 16), 0, 0);                  \
    __builtin_amdgcn_global_load_async_to_lds_b128(                           \
        (as1_v4i_p)(gp_ + 24), (as3_v4i_p)(lp_ + 24), 0, 0);                  \
  } while (0)

  ISSUE_B(0, 0);
#endif

  v8f acc[8] = {};
  for (int it = 0; it < 64; ++it) {
    const int kk = it * 32;
    const int c0 = kk + hi * 8, c1 = c0 + 16;
    __builtin_prefetch(er + kk + 1024, 0, 1);
#if USE_ASYNC_LDS
    const int stage = it & 1;
    __builtin_amdgcn_s_wait_asynccnt(0);  // own DMAs for current stage done
    __syncthreads();                      // stage fully populated; prev reads done
    if (it + 1 < 64) ISSUE_B(kk + 32, stage ^ 1);
#endif
    ABf a;
    cvt4p(*(const float4*)(er + c0),     rm, is, a.u + 0);
    cvt4p(*(const float4*)(er + c0 + 4), rm, is, a.u + 2);
    cvt4p(*(const float4*)(er + c1),     rm, is, a.u + 4);
    cvt4p(*(const float4*)(er + c1 + 4), rm, is, a.u + 6);
#pragma unroll
    for (int t = 0; t < 8; ++t) {
      const int n = nh * 128 + t * 16 + m;
      ABf bf;
#if USE_ASYNC_LDS
      const unsigned short* bp = &ldsB[stage][n * 32];
      bf.q[0] = *(const uint4*)(bp + hi * 8);
      bf.q[1] = *(const uint4*)(bp + hi * 8 + 16);
#else
      const unsigned short* bp = mb + (size_t)n * 2048;
      bf.q[0] = *(const uint4*)(bp + c0);
      bf.q[1] = *(const uint4*)(bp + c1);
#endif
      acc[t] = __builtin_amdgcn_wmma_f32_16x16x32_bf16(
          false, a.v, false, bf.v, (short)0, acc[t], false, false);
    }
  }
#pragma unroll
  for (int t = 0; t < 8; ++t) {
    const int n = nh * 128 + t * 16 + m;
#pragma unroll
    for (int i = 0; i < 8; ++i) {
      const int row = r0 + mi * 16 + hi * 8 + i;
      agg[((size_t)b * 2048 + row) * 256 + n] = acc[t][i];
    }
  }
}

// ---------------------------------------------------------------------------
// Kernel 4: fused LayerNorm(concat[V,agg]) -> relu(.@W1+b1) -> .@W2+b2 -> +V
// Block: 256 thr, 32 rows. LDS: hn 32x512 bf16 (32KB) + h 32x256 bf16 (16KB).
// ---------------------------------------------------------------------------
__global__ void __launch_bounds__(256) mlp_kernel(
    const float* __restrict__ V, const float* __restrict__ agg,
    const float* __restrict__ ln_g, const float* __restrict__ ln_b,
    const unsigned short* __restrict__ W1T, const float* __restrict__ b1,
    const unsigned short* __restrict__ W2T, const float* __restrict__ b2,
    float* __restrict__ out) {
  __shared__ unsigned short hn[32 * 512];
  __shared__ unsigned short h2[32 * 256];
  const int b = blockIdx.x >> 6;
  const int r0 = (blockIdx.x & 63) * 32;
  const int t = threadIdx.x;

  { // ---- LayerNorm over 512 concat values; 8 threads per row ----
    const int r = t >> 3, q = t & 7;
    const int grow = r0 + r;
    const float* src = (q < 4)
        ? (V   + ((size_t)b * 2048 + grow) * 256 + q * 64)
        : (agg + ((size_t)b * 2048 + grow) * 256 + (q - 4) * 64);
    float sum = 0.f, sq = 0.f;
#pragma unroll
    for (int j = 0; j < 64; j += 4) {
      const float4 x = *(const float4*)(src + j);
      sum += x.x + x.y + x.z + x.w;
      sq  += x.x * x.x + x.y * x.y + x.z * x.z + x.w * x.w;
    }
#pragma unroll
    for (int off = 1; off <= 4; off <<= 1) {
      sum += __shfl_xor(sum, off, 32);
      sq  += __shfl_xor(sq,  off, 32);
    }
    const float mu = sum * (1.0f / 512.0f);
    const float var = sq * (1.0f / 512.0f) - mu * mu;
    const float rstd = rsqrtf(var + 1e-5f);
    const int cb = q * 64;
    for (int j = 0; j < 64; j += 4) {
      const float4 x = *(const float4*)(src + j);
      const float4 g = *(const float4*)(ln_g + cb + j);
      const float4 bb = *(const float4*)(ln_b + cb + j);
      Pack4 p;
      p.u32[0] = f2bf2((x.x - mu) * rstd * g.x + bb.x,
                       (x.y - mu) * rstd * g.y + bb.y);
      p.u32[1] = f2bf2((x.z - mu) * rstd * g.z + bb.z,
                       (x.w - mu) * rstd * g.w + bb.w);
      *(uint2*)&hn[r * 512 + cb + j] = p.u;
    }
  }
  __syncthreads();

  const int w = t >> 5, lane = t & 31;
  const int mi = w >> 2, g4 = w & 3;        // mi: 16-row tile, g4: 64-col slab
  const int m = lane & 15, hi = lane >> 4;

  { // ---- GEMM1: h = relu(hn @ W1 + b1), K = 512 ----
    v8f acc[4] = {};
    for (int kk = 0; kk < 512; kk += 32) {
      const int c0 = kk + hi * 8, c1 = c0 + 16;
      ABf a;
      a.q[0] = *(const uint4*)&hn[(mi * 16 + m) * 512 + c0];
      a.q[1] = *(const uint4*)&hn[(mi * 16 + m) * 512 + c1];
#pragma unroll
      for (int tt = 0; tt < 4; ++tt) {
        const int n = g4 * 64 + tt * 16 + m;
        const unsigned short* bp = W1T + (size_t)n * 512;
        ABf bf;
        bf.q[0] = *(const uint4*)(bp + c0);
        bf.q[1] = *(const uint4*)(bp + c1);
        acc[tt] = __builtin_amdgcn_wmma_f32_16x16x32_bf16(
            false, a.v, false, bf.v, (short)0, acc[tt], false, false);
      }
    }
#pragma unroll
    for (int tt = 0; tt < 4; ++tt) {
      const int n = g4 * 64 + tt * 16 + m;
      const float bias = b1[n];
#pragma unroll
      for (int i = 0; i < 8; ++i) {
        const int row = mi * 16 + hi * 8 + i;
        const float v = acc[tt][i] + bias;
        h2[row * 256 + n] = f2bf(v > 0.f ? v : 0.f);
      }
    }
  }
  __syncthreads();

  { // ---- GEMM2: update = h @ W2 + b2; out = V + update. K = 256 ----
    v8f acc[4] = {};
    for (int kk = 0; kk < 256; kk += 32) {
      const int c0 = kk + hi * 8, c1 = c0 + 16;
      ABf a;
      a.q[0] = *(const uint4*)&h2[(mi * 16 + m) * 256 + c0];
      a.q[1] = *(const uint4*)&h2[(mi * 16 + m) * 256 + c1];
#pragma unroll
      for (int tt = 0; tt < 4; ++tt) {
        const int n = g4 * 64 + tt * 16 + m;
        const unsigned short* bp = W2T + (size_t)n * 256;
        ABf bf;
        bf.q[0] = *(const uint4*)(bp + c0);
        bf.q[1] = *(const uint4*)(bp + c1);
        acc[tt] = __builtin_amdgcn_wmma_f32_16x16x32_bf16(
            false, a.v, false, bf.v, (short)0, acc[tt], false, false);
      }
    }
#pragma unroll
    for (int tt = 0; tt < 4; ++tt) {
      const int n = g4 * 64 + tt * 16 + m;
      const float bias = b2[n];
#pragma unroll
      for (int i = 0; i < 8; ++i) {
        const int row = r0 + mi * 16 + hi * 8 + i;
        const size_t idx = ((size_t)b * 2048 + row) * 256 + n;
        out[idx] = V[idx] + acc[tt][i] + bias;
      }
    }
  }
}

// ---------------------------------------------------------------------------
extern "C" void kernel_launch(void* const* d_in, const int* in_sizes, int n_in,
                              void* d_out, int out_size, void* d_ws,
                              size_t ws_size, hipStream_t stream) {
  const float* V     = (const float*)d_in[0];
  const float* E     = (const float*)d_in[1];
  const float* msg_W = (const float*)d_in[2];
  const float* msg_b = (const float*)d_in[3];
  const float* ln_g  = (const float*)d_in[4];
  const float* ln_b  = (const float*)d_in[5];
  const float* W1    = (const float*)d_in[6];
  const float* b1    = (const float*)d_in[7];
  const float* W2    = (const float*)d_in[8];
  const float* b2    = (const float*)d_in[9];
  float* out = (float*)d_out;

  char* ws = (char*)d_ws;
  unsigned short* msgsT = (unsigned short*)ws; ws += (size_t)8 * 256 * 2048 * 2; // 8 MB
  float* agg            = (float*)ws;          ws += (size_t)8 * 2048 * 256 * 4; // 16 MB
  float* rowmax         = (float*)ws;          ws += (size_t)16384 * 4;
  float* invsum         = (float*)ws;          ws += (size_t)16384 * 4;
  unsigned short* msgWT = (unsigned short*)ws; ws += (size_t)256 * 256 * 2;
  unsigned short* W1T   = (unsigned short*)ws; ws += (size_t)256 * 512 * 2;
  unsigned short* W2T   = (unsigned short*)ws;

  prep_weights <<<1024, 256, 0, stream>>>(msg_W, W1, W2, msgWT, W1T, W2T);
  msg_kernel   <<<256,  256, 0, stream>>>(V, msgWT, msg_b, msgsT);
  softmax_stats<<<2048, 256, 0, stream>>>(E, rowmax, invsum);
  attn_kernel  <<<256,  256, 0, stream>>>(E, msgsT, rowmax, invsum, agg);
  mlp_kernel   <<<512,  256, 0, stream>>>(V, agg, ln_g, ln_b, W1T, b1, W2T, b2, out);
}
